// CrossAttention_48859547959652
// MI455X (gfx1250) — compile-verified, matrix operand order checked by reading, not run on hardware
//
#include <hip/hip_runtime.h>
#include <hip/hip_bf16.h>

// ---------------------------------------------------------------------------
// CDNA5 (gfx1250) cross-attention forward, v2.
// - One-time f32->bf16 conversion of activations + weights (bandwidth trivial)
// - Pure-bf16 WMMA GEMMs with 32x64 per-wave tiles (8 WMMAs / k-step)
// - Flash attention with online softmax, per-wave LDS P-transpose
// ---------------------------------------------------------------------------

typedef __bf16 bf16_t;
typedef __attribute__((ext_vector_type(16))) __bf16 v16bf;
typedef __attribute__((ext_vector_type(8)))  __bf16 v8bf;
typedef __attribute__((ext_vector_type(8)))  float  v8f;

#define DIM_C   1024
#define HEADS_C 16
#define HD_C    64
#define B_C     2
#define L_C     2048
#define K_C     2048

__device__ __forceinline__ v8f v8f_zero() {
    v8f z;
#pragma unroll
    for (int i = 0; i < 8; ++i) z[i] = 0.0f;
    return z;
}

__device__ __forceinline__ v8f wmma_bf16(v16bf a, v16bf b, v8f c) {
    return __builtin_amdgcn_wmma_f32_16x16x32_bf16(
        /*neg_a=*/false, a, /*neg_b=*/false, b,
        /*c_mod=*/(short)0, c, /*reuse_a=*/false, /*reuse_b=*/false);
}

// A/B fragment for one 32-wide K slice, ISA 16-bit layout:
// lane&15 selects row (A) / column (B); kbase = (lane>>4)*8.
// frag[0..7]  <- elems kbase..kbase+7
// frag[8..15] <- elems 16+kbase..16+kbase+7
__device__ __forceinline__ v16bf frag_from_bf16(const bf16_t* base, int kbase) {
    v8bf lo = *(const v8bf*)(base + kbase);
    v8bf hi = *(const v8bf*)(base + 16 + kbase);
    v16bf f;
#pragma unroll
    for (int i = 0; i < 8; ++i) { f[i] = lo[i]; f[8 + i] = hi[i]; }
    return f;
}

// ---------------------------------------------------------------------------
// Bulk f32 -> bf16 conversion (8 elems / thread, b128 in, b128 out)
// ---------------------------------------------------------------------------
__global__ void cvt_f32_bf16_kernel(const float* __restrict__ x,
                                    bf16_t* __restrict__ y, int n8) {
    const int i = blockIdx.x * blockDim.x + threadIdx.x;
    if (i >= n8) return;
    const float4 a = ((const float4*)x)[2 * i];
    const float4 b = ((const float4*)x)[2 * i + 1];
    v8bf o;
    o[0] = (__bf16)a.x; o[1] = (__bf16)a.y; o[2] = (__bf16)a.z; o[3] = (__bf16)a.w;
    o[4] = (__bf16)b.x; o[5] = (__bf16)b.y; o[6] = (__bf16)b.z; o[7] = (__bf16)b.w;
    ((v8bf*)y)[i] = o;
}

// ---------------------------------------------------------------------------
// GEMM core: one wave computes a 32x64 tile of Y = X @ W^T (all bf16 in).
// X: [M,1024] bf16 row-major, W: [1024,1024] bf16 row-major.
// acc[half][t]: rows mt*32 + half*16 + ..., cols nb*64 + t*16 + ...
// ---------------------------------------------------------------------------
__device__ __forceinline__ void gemm_core_32x64(const bf16_t* __restrict__ X,
                                                const bf16_t* __restrict__ W,
                                                int mt, int nb, int lane,
                                                v8f acc[2][4]) {
    const int col   = lane & 15;
    const int kbase = (lane >> 4) * 8;
    const bf16_t* xrow0 = X + (size_t)(mt * 32 + col) * DIM_C;
    const bf16_t* xrow1 = X + (size_t)(mt * 32 + 16 + col) * DIM_C;
    const bf16_t* wrow  = W + (size_t)(nb * 64 + col) * DIM_C;
    for (int kt = 0; kt < DIM_C; kt += 32) {
        // prefetch next A k-tiles (gfx1250 global_prefetch path)
        __builtin_prefetch(xrow0 + kt + 32, 0, 0);
        __builtin_prefetch(xrow1 + kt + 32, 0, 0);
        v16bf a0 = frag_from_bf16(xrow0 + kt, kbase);
        v16bf a1 = frag_from_bf16(xrow1 + kt, kbase);
#pragma unroll
        for (int t = 0; t < 4; ++t) {
            v16bf b = frag_from_bf16(wrow + (size_t)t * 16 * DIM_C + kt, kbase);
            acc[0][t] = wmma_bf16(a0, b, acc[0][t]);
            acc[1][t] = wmma_bf16(a1, b, acc[1][t]);
        }
    }
}

// Q/K/V projection: Y = X@W^T + b, emitted as bf16.
// vtrans==0 -> out[((b*H+h)*S + s)*64 + d]   ([B,H,S,D])
// vtrans==1 -> out[((b*H+h)*64 + d)*S + s]   ([B,H,D,S], for V)
__global__ void __launch_bounds__(128)
proj_qkv_kernel(const bf16_t* __restrict__ X, const bf16_t* __restrict__ W,
                const float* __restrict__ bias, bf16_t* __restrict__ out,
                int S, int vtrans) {
    const int lane = threadIdx.x & 31;
    const int wid  = threadIdx.x >> 5;
    const int gw   = blockIdx.x * 4 + wid;     // 2048 waves
    const int mt   = gw >> 4;                  // 128 row tiles of 32
    const int nb   = gw & 15;                  // 16 column blocks of 64

    v8f acc[2][4];
#pragma unroll
    for (int hh = 0; hh < 2; ++hh)
#pragma unroll
        for (int t = 0; t < 4; ++t) acc[hh][t] = v8f_zero();
    gemm_core_32x64(X, W, mt, nb, lane, acc);

    const int col = lane & 15;
    const int mo  = (lane >> 4) * 8;
#pragma unroll
    for (int t = 0; t < 4; ++t) {
        const int n  = nb * 64 + t * 16 + col;
        const float bv = bias[n];
        const int h = n >> 6, d = n & 63;
#pragma unroll
        for (int hh = 0; hh < 2; ++hh) {
#pragma unroll
            for (int r = 0; r < 8; ++r) {
                const int m = mt * 32 + hh * 16 + mo + r;
                const int b = m / S, s = m % S;
                const size_t o = vtrans
                    ? (((size_t)(b * HEADS_C + h) * 64 + d) * (size_t)S + s)
                    : (((size_t)(b * HEADS_C + h) * (size_t)S + s) * 64 + d);
                out[o] = (__bf16)(acc[hh][t][r] + bv);
            }
        }
    }
}

// Output projection: Y = X@Wp^T + bp, times target mask, fp32 out [M,1024].
__global__ void __launch_bounds__(128)
proj_out_kernel(const bf16_t* __restrict__ X, const bf16_t* __restrict__ W,
                const float* __restrict__ bias,
                const unsigned char* __restrict__ tmask,
                float* __restrict__ out) {
    const int lane = threadIdx.x & 31;
    const int wid  = threadIdx.x >> 5;
    const int gw   = blockIdx.x * 4 + wid;
    const int mt   = gw >> 4;
    const int nb   = gw & 15;

    v8f acc[2][4];
#pragma unroll
    for (int hh = 0; hh < 2; ++hh)
#pragma unroll
        for (int t = 0; t < 4; ++t) acc[hh][t] = v8f_zero();
    gemm_core_32x64(X, W, mt, nb, lane, acc);

    const int col = lane & 15;
    const int mo  = (lane >> 4) * 8;
#pragma unroll
    for (int t = 0; t < 4; ++t) {
        const int n  = nb * 64 + t * 16 + col;
        const float bv = bias[n];
#pragma unroll
        for (int hh = 0; hh < 2; ++hh) {
#pragma unroll
            for (int r = 0; r < 8; ++r) {
                const int m = mt * 32 + hh * 16 + mo + r;
                const float v = acc[hh][t][r] + bv;
                out[(size_t)m * DIM_C + n] = tmask[m] ? v : 0.0f;
            }
        }
    }
}

// ---------------------------------------------------------------------------
// RoPE, in place on bf16 [BH, S, 64]. One thread owns the (i, i+32) pair.
// ---------------------------------------------------------------------------
__global__ void rope_kernel(bf16_t* __restrict__ x, int S) {
    const int total = B_C * HEADS_C * S * 32;
    const int idx = blockIdx.x * blockDim.x + threadIdx.x;
    if (idx >= total) return;
    const int i  = idx & 31;
    const int s  = (idx >> 5) % S;
    const int bh = idx / (32 * S);
    // inv_freq[i] = 10000^(-2i/64)
    const float inv = __expf((-2.0f * (float)i / 64.0f) * 9.2103403719761836f);
    const float ang = (float)s * inv;
    const float c = __cosf(ang), sn = __sinf(ang);
    bf16_t* p = x + ((size_t)bh * S + s) * 64;
    const float x1 = (float)p[i];
    const float x2 = (float)p[i + 32];
    p[i]      = (__bf16)(x1 * c - x2 * sn);
    p[i + 32] = (__bf16)(x2 * c + x1 * sn);
}

// ---------------------------------------------------------------------------
// Flash attention: one wave per (b, h, 16-row q tile); online softmax.
// Q,K: [B,H,S,64] bf16;  V: [B,H,64,K] bf16 (transposed);
// out: [B,L,1024] bf16 (feeds the bf16 output-projection GEMM).
// ---------------------------------------------------------------------------
__global__ void __launch_bounds__(128)
attn_kernel(const bf16_t* __restrict__ Qb, const bf16_t* __restrict__ Kb,
            const bf16_t* __restrict__ Vt, const unsigned char* __restrict__ cmask,
            bf16_t* __restrict__ attnout) {
    __shared__ __align__(16) bf16_t Pst[4][16 * 32];   // per-wave P staging

    const int lane = threadIdx.x & 31;
    const int wid  = threadIdx.x >> 5;
    const int gw   = blockIdx.x * 4 + wid;
    const int qt   = gw % (L_C / 16);
    const int bh   = gw / (L_C / 16);
    const int b    = bh >> 4;
    const int h    = bh & 15;

    const bf16_t* Qh = Qb + (size_t)bh * L_C * HD_C;
    const bf16_t* Kh = Kb + (size_t)bh * K_C * HD_C;
    const bf16_t* Vh = Vt + (size_t)bh * HD_C * K_C;
    const unsigned char* cm = cmask + (size_t)b * K_C;

    const int col   = lane & 15;
    const int kbase = (lane >> 4) * 8;
    const int mo    = (lane >> 4) * 8;

    const bf16_t* qrow = Qh + (size_t)(qt * 16 + col) * HD_C;
    const v16bf q0 = frag_from_bf16(qrow, kbase);        // d = 0..31
    const v16bf q1 = frag_from_bf16(qrow + 32, kbase);   // d = 32..63

    v8f acc[4];
#pragma unroll
    for (int t = 0; t < 4; ++t) acc[t] = v8f_zero();
    float rmax[8], rsum[8];
#pragma unroll
    for (int r = 0; r < 8; ++r) { rmax[r] = -1e30f; rsum[r] = 0.0f; }

    bf16_t* P = Pst[wid];
    const float scale = 0.125f;   // 64^-0.5

    for (int kc = 0; kc < K_C; kc += 32) {
        float p[2][8];
#pragma unroll
        for (int sub = 0; sub < 2; ++sub) {
            const int key = kc + sub * 16 + col;
            const bf16_t* krow = Kh + (size_t)key * HD_C;
            v16bf k0 = frag_from_bf16(krow, kbase);
            v16bf k1 = frag_from_bf16(krow + 32, kbase);
            v8f s = v8f_zero();
            s = wmma_bf16(q0, k0, s);
            s = wmma_bf16(q1, k1, s);
            const float mval = cm[key] ? 0.0f : -1e30f;
#pragma unroll
            for (int r = 0; r < 8; ++r) p[sub][r] = s[r] * scale + mval;
        }
        // online softmax per output row (16-lane group reductions)
#pragma unroll
        for (int r = 0; r < 8; ++r) {
            float v = fmaxf(p[0][r], p[1][r]);
            v = fmaxf(v, __shfl_xor(v, 1));
            v = fmaxf(v, __shfl_xor(v, 2));
            v = fmaxf(v, __shfl_xor(v, 4));
            v = fmaxf(v, __shfl_xor(v, 8));
            const float mnew = fmaxf(rmax[r], v);
            const float fs = __expf(rmax[r] - mnew);
            rmax[r] = mnew;
            rsum[r] *= fs;
#pragma unroll
            for (int t = 0; t < 4; ++t) acc[t][r] *= fs;
            const float e0 = __expf(p[0][r] - mnew);
            const float e1 = __expf(p[1][r] - mnew);
            P[(mo + r) * 32 + col]      = (__bf16)e0;
            P[(mo + r) * 32 + 16 + col] = (__bf16)e1;
            float sm = e0 + e1;
            sm += __shfl_xor(sm, 1);
            sm += __shfl_xor(sm, 2);
            sm += __shfl_xor(sm, 4);
            sm += __shfl_xor(sm, 8);
            rsum[r] += sm;
        }
        // P (C layout) -> A fragment via per-wave LDS tile
        v16bf pf = frag_from_bf16(P + col * 32, kbase);
#pragma unroll
        for (int t = 0; t < 4; ++t) {
            const bf16_t* vcol = Vh + (size_t)(t * 16 + col) * K_C + kc;
            v16bf vf = frag_from_bf16(vcol, kbase);
            acc[t] = wmma_bf16(pf, vf, acc[t]);
        }
    }

#pragma unroll
    for (int r = 0; r < 8; ++r) {
        const float inv = __fdividef(1.0f, rsum[r]);
        const int m = qt * 16 + mo + r;
#pragma unroll
        for (int t = 0; t < 4; ++t) {
            const int d = t * 16 + col;
            attnout[((size_t)(b * L_C + m)) * DIM_C + h * 64 + d] =
                (__bf16)(acc[t][r] * inv);
        }
    }
}

// ---------------------------------------------------------------------------
extern "C" void kernel_launch(void* const* d_in, const int* in_sizes, int n_in,
                              void* d_out, int out_size, void* d_ws, size_t ws_size,
                              hipStream_t stream) {
    (void)in_sizes; (void)n_in; (void)out_size; (void)ws_size;

    const float* target  = (const float*)d_in[0];
    const float* context = (const float*)d_in[1];
    const unsigned char* tmask = (const unsigned char*)d_in[2];
    const unsigned char* cmask = (const unsigned char*)d_in[3];
    const float* Wq = (const float*)d_in[4];
    const float* bq = (const float*)d_in[5];
    const float* Wk = (const float*)d_in[6];
    const float* bk = (const float*)d_in[7];
    const float* Wv = (const float*)d_in[8];
    const float* bv = (const float*)d_in[9];
    const float* Wp = (const float*)d_in[10];
    const float* bp = (const float*)d_in[11];

    // Workspace layout (all bf16 unless noted), sizes in elements:
    const size_t nAct = (size_t)B_C * L_C * DIM_C;        // 4,194,304
    const size_t nW   = (size_t)DIM_C * DIM_C;            // 1,048,576
    char* ws = (char*)d_ws;
    size_t off = 0;
    bf16_t* Tb  = (bf16_t*)(ws + off); off += nAct * 2;   // target bf16
    bf16_t* Cb  = (bf16_t*)(ws + off); off += nAct * 2;   // context bf16
    bf16_t* Wqb = (bf16_t*)(ws + off); off += nW * 2;
    bf16_t* Wkb = (bf16_t*)(ws + off); off += nW * 2;
    bf16_t* Wvb = (bf16_t*)(ws + off); off += nW * 2;
    bf16_t* Wpb = (bf16_t*)(ws + off); off += nW * 2;
    bf16_t* Qb  = (bf16_t*)(ws + off); off += nAct * 2;   // [B,H,L,64]
    bf16_t* Kb  = (bf16_t*)(ws + off); off += nAct * 2;   // [B,H,K,64]
    bf16_t* Vt  = (bf16_t*)(ws + off); off += nAct * 2;   // [B,H,64,K]
    bf16_t* att = (bf16_t*)(ws + off); off += nAct * 2;   // [B,L,1024]

    // One-time f32 -> bf16 conversions (bandwidth-trivial)
    const int actN8 = (int)(nAct / 8), wN8 = (int)(nW / 8);
    cvt_f32_bf16_kernel<<<(actN8 + 255) / 256, 256, 0, stream>>>(target,  Tb,  actN8);
    cvt_f32_bf16_kernel<<<(actN8 + 255) / 256, 256, 0, stream>>>(context, Cb,  actN8);
    cvt_f32_bf16_kernel<<<(wN8 + 255) / 256,   256, 0, stream>>>(Wq, Wqb, wN8);
    cvt_f32_bf16_kernel<<<(wN8 + 255) / 256,   256, 0, stream>>>(Wk, Wkb, wN8);
    cvt_f32_bf16_kernel<<<(wN8 + 255) / 256,   256, 0, stream>>>(Wv, Wvb, wN8);
    cvt_f32_bf16_kernel<<<(wN8 + 255) / 256,   256, 0, stream>>>(Wp, Wpb, wN8);

    // Projections: 2048 waves each -> 512 blocks of 4 waves
    proj_qkv_kernel<<<512, 128, 0, stream>>>(Tb, Wqb, bq, Qb, L_C, 0);
    proj_qkv_kernel<<<512, 128, 0, stream>>>(Cb, Wkb, bk, Kb, K_C, 0);
    proj_qkv_kernel<<<512, 128, 0, stream>>>(Cb, Wvb, bv, Vt, K_C, 1);

    const int ropeThreads = B_C * HEADS_C * L_C * 32;
    rope_kernel<<<(ropeThreads + 255) / 256, 256, 0, stream>>>(Qb, L_C);
    rope_kernel<<<(ropeThreads + 255) / 256, 256, 0, stream>>>(Kb, K_C);

    attn_kernel<<<1024, 128, 0, stream>>>(Qb, Kb, Vt, cmask, att);

    proj_out_kernel<<<512, 128, 0, stream>>>(att, Wpb, bp, tmask, (float*)d_out);
}
